// SelfAttention_85177791414312
// MI455X (gfx1250) — matrix operand order repaired and checked by reading, not compiled, not run on hardware
//
#include <hip/hip_runtime.h>
#include <hip/hip_bf16.h>
#include <math.h>

typedef _Float16 f16;
typedef __attribute__((ext_vector_type(16))) _Float16 v16h;
typedef __attribute__((ext_vector_type(8)))  _Float16 v8h;
typedef __attribute__((ext_vector_type(8)))  float    v8f;
typedef __attribute__((ext_vector_type(4)))  unsigned int u32x4;
typedef __attribute__((ext_vector_type(8)))  int i32x8;
typedef __attribute__((ext_vector_type(4)))  int i32x4;

#define HID   768
#define NH    12
#define HD    64
#define BATCH 2
#define SEQ   4096
#define MTOT  (BATCH*SEQ)                 // 8192 tokens
#define XE    ((size_t)MTOT * HID)        // elements in one [tokens, HID] tensor

static __device__ __forceinline__ v16h cat8(v8h a, v8h b) {
  return __builtin_shufflevector(a, b, 0,1,2,3,4,5,6,7,8,9,10,11,12,13,14,15);
}

static __device__ __forceinline__ v8f wmma16(v16h a, v16h b, v8f c) {
  // D = A(16x32 f16) * B(32x16 f16) + C(16x16 f32)
  return __builtin_amdgcn_wmma_f32_16x16x32_f16(false, a, false, b, (short)0, c, false, false);
}

// -------------------------------------------------------------- TDM staging
// 2D-tile D# (ISA 8.3/8.4) -> TENSOR_LOAD_TO_LDS. data_size = 2 bytes.
// pad: pad_interval=32 DWORDs (code 4), pad_amount=4 DWORDs (code 3)
//      -> LDS row stride = tile0 + 8 halves (64 -> 72).
static __device__ __forceinline__ void tdm_load_2d(unsigned int lds_byte_off,
                                                   const void* gaddr,
                                                   unsigned int tile0,   // contiguous dim, halves
                                                   unsigned int tile1,   // rows
                                                   unsigned long long stride0, // row stride, halves
                                                   int pad) {
  unsigned long long ga = (unsigned long long)(uintptr_t)gaddr;
  u32x4 g0;
  g0[0] = 1u;                                       // count=1 (valid user D#)
  g0[1] = lds_byte_off;                             // lds_addr
  g0[2] = (unsigned int)ga;                         // global_addr[31:0]
  g0[3] = (unsigned int)((ga >> 32) & 0x01FFFFFFu)  // global_addr[56:32]
        | (2u << 30);                               // type=2 ("image")
  unsigned int w0 = (1u << 16);                     // data_size=1 -> 2 bytes
  if (pad) w0 |= (1u << 20) | (4u << 22) | (3u << 25);
  i32x8 g1;
  g1[0] = (int)w0;
  g1[1] = (int)((tile0 & 0xFFFFu) << 16);           // tensor_dim0[15:0]
  g1[2] = (int)(((tile0 >> 16) & 0xFFFFu) | ((tile1 & 0xFFFFu) << 16)); // td0 hi | td1 lo
  g1[3] = (int)(((tile1 >> 16) & 0xFFFFu) | ((tile0 & 0xFFFFu) << 16)); // td1 hi | tile_dim0
  g1[4] = (int)(tile1 & 0xFFFFu);                   // tile_dim1 | tile_dim2=0
  g1[5] = (int)(unsigned int)stride0;               // tensor_dim0_stride[31:0]
  g1[6] = (int)(unsigned int)((stride0 >> 32) & 0xFFFFu);
  g1[7] = 0;
  i32x4 z4 = {0, 0, 0, 0};
#if __clang_major__ >= 23
  i32x8 z8 = {0, 0, 0, 0, 0, 0, 0, 0};
  __builtin_amdgcn_tensor_load_to_lds(g0, g1, z4, z4, z8, 0);
#else
  __builtin_amdgcn_tensor_load_to_lds(g0, g1, z4, z4, 0);
#endif
}

static __device__ __forceinline__ unsigned int lds_off_of(const void* p) {
  return (unsigned int)(uintptr_t)p;   // low 32 bits of generic LDS address = LDS offset
}

// ---------------------------------------------------------------- fp32 -> f16
__global__ void cvt_f32_to_f16(const float* __restrict__ src, f16* __restrict__ dst, int n) {
  int i = blockIdx.x * blockDim.x + threadIdx.x;
  int stride = gridDim.x * blockDim.x;
  for (; i < n; i += stride) dst[i] = (f16)src[i];
}

// ------------------------------------------- fused transpose + fp32 -> f16
__global__ void __launch_bounds__(256)
transpose_w(const float* __restrict__ W, f16* __restrict__ Wt, int rowOffset) {
  __shared__ float tile[32][33];
  const int tx = threadIdx.x;
  const int ty = threadIdx.y;
  const int x  = blockIdx.x * 32 + tx;
  const int y0 = blockIdx.y * 32;
  #pragma unroll
  for (int j = ty; j < 32; j += 8)
    tile[j][tx] = W[(size_t)(y0 + j) * HID + x];
  __syncthreads();
  const int n0 = blockIdx.x * 32;
  const int xo = y0 + tx;
  #pragma unroll
  for (int j = ty; j < 32; j += 8)
    Wt[(size_t)(rowOffset + n0 + j) * HID + xo] = (f16)tile[tx][j];
}

// ---------------------------------------------------------------- tiled GEMM
// Y[M=8192, N] = A[M,768] @ WtT + bias,  Wt stored transposed [N][768] f16.
// MODE 0: N=2304 packed Q|K|V. Q,K -> f16 head-split [bh][tok][d];
//         V -> f16 head-split TRANSPOSED [bh][d][tok] (feeds TDM in attention).
// MODE 1: N=768 (Wo) -> f32 resid = Y + hidden.
// 256 threads (8 waves). Tile 128(M) x 64(N), K step 64.
// A/B tiles staged by double-buffered Tensor Data Mover loads (HW pad, stride 72).
#define GS 72
template <int MODE>
__global__ void __launch_bounds__(256)
gemm768(const f16* __restrict__ A, const f16* __restrict__ Wt,
        const float* __restrict__ b0p, const float* __restrict__ b1p,
        const float* __restrict__ b2p, const float* __restrict__ hidden,
        f16* __restrict__ dst16, float* __restrict__ dst32) {
  __shared__ __align__(16) f16 As[2][128 * GS];   // 2 x 18 KB
  __shared__ __align__(16) f16 Bs[2][64 * GS];    // 2 x  9 KB

  const int t    = threadIdx.x;
  const int m0g  = blockIdx.x * 128;
  const int n0g  = blockIdx.y * 64;
  const int wave = t >> 5;
  const int lane = t & 31;
  const int lo   = lane & 15;
  const int hi   = lane >> 4;
  const int wm   = (wave >> 1) * 32;
  const int wn   = (wave & 1) * 32;

  v8f acc[2][2] = {};

  if (wave == 0) {
    tdm_load_2d(lds_off_of(&As[0][0]), &A[(size_t)m0g * HID], 64, 128, HID, 1);
    tdm_load_2d(lds_off_of(&Bs[0][0]), &Wt[(size_t)n0g * HID], 64, 64, HID, 1);
    __builtin_amdgcn_s_wait_tensorcnt(0);
  }
  __syncthreads();

  int cur = 0;
  for (int kb = 0; kb < HID; kb += 64, cur ^= 1) {
    // Prefetch next tiles into the other buffer while we compute.
    if (wave == 0 && kb + 64 < HID) {
      tdm_load_2d(lds_off_of(&As[cur ^ 1][0]), &A[(size_t)m0g * HID + kb + 64], 64, 128, HID, 1);
      tdm_load_2d(lds_off_of(&Bs[cur ^ 1][0]), &Wt[(size_t)n0g * HID + kb + 64], 64, 64, HID, 1);
    }

    v16h a[2][2], b[2][2];          // [k-chunk][tile]
    #pragma unroll
    for (int c = 0; c < 2; ++c) {
      #pragma unroll
      for (int mi = 0; mi < 2; ++mi) {
        int row = wm + mi * 16 + lo;
        const f16* ap = &As[cur][row * GS + c * 32 + hi * 16];
        a[c][mi] = cat8(*(const v8h*)ap, *(const v8h*)(ap + 8));
      }
      #pragma unroll
      for (int ni = 0; ni < 2; ++ni) {
        int n = wn + ni * 16 + lo;
        const f16* bp = &Bs[cur][n * GS + c * 32 + hi * 16];
        b[c][ni] = cat8(*(const v8h*)bp, *(const v8h*)(bp + 8));
      }
    }
    #pragma unroll
    for (int c = 0; c < 2; ++c)
      #pragma unroll
      for (int mi = 0; mi < 2; ++mi)
        #pragma unroll
        for (int ni = 0; ni < 2; ++ni)
          acc[mi][ni] = wmma16(a[c][mi], b[c][ni], acc[mi][ni]);

    if (wave == 0) __builtin_amdgcn_s_wait_tensorcnt(0);
    __syncthreads();   // next buffer ready for everyone; current buffer free
  }

  // Epilogue. D layout: element e -> row (e + 8*hi), col = lo.
  #pragma unroll
  for (int mi = 0; mi < 2; ++mi)
    #pragma unroll
    for (int ni = 0; ni < 2; ++ni)
      #pragma unroll
      for (int e = 0; e < 8; ++e) {
        int rowg = m0g + wm + mi * 16 + e + 8 * hi;
        int colg = n0g + wn + ni * 16 + lo;
        if (MODE == 0) {
          int proj = colg / HID;              // 0=Q 1=K 2=V
          int c    = colg - proj * HID;
          const float* bp = (proj == 0) ? b0p : (proj == 1) ? b1p : b2p;
          float v  = acc[mi][ni][e] + bp[c];
          int bidx = rowg >> 12;              // rowg / SEQ
          int s    = rowg & (SEQ - 1);
          int hcol = c >> 6;                  // head
          int d    = c & 63;
          size_t bhBase = ((size_t)bidx * NH + hcol) * SEQ;   // (b*NH+h)*SEQ
          if (proj == 2)  // V transposed: [bh][d][token]
            dst16[2 * XE + bhBase * HD + (size_t)d * SEQ + s] = (f16)v;
          else            // Q, K row-major: [bh][token][d]
            dst16[(size_t)proj * XE + (bhBase + s) * HD + d] = (f16)v;
        } else {
          float v = acc[mi][ni][e] + b0p[colg];
          size_t idx = (size_t)rowg * HID + colg;
          dst32[idx] = v + hidden[idx];
        }
      }
}

// ---------------------------------------------------------------- attention
// 8 waves per workgroup; 128 query rows of one (b, h) per workgroup.
// K [bh][tok][d] and V-transposed [bh][d][tok] chunks staged by
// double-buffered TDM loads (HW-padded LDS rows, stride 72).
#define KT 64                                  // keys per step
#define KS 72                                  // padded LDS row stride (halves)
__global__ void __launch_bounds__(256)
attn_kernel(const f16* __restrict__ Q, const f16* __restrict__ K,
            const f16* __restrict__ V, const float* __restrict__ mask,
            f16* __restrict__ ctx) {
  __shared__ __align__(16) f16 Ks[2][KT * KS];     // [tok][d]
  __shared__ __align__(16) f16 Vt[2][HD * KS];     // [d][tok]
  __shared__ __align__(16) f16 Pt[8 * 16 * KT];    // per-wave P tiles

  const int bh   = blockIdx.x;
  const int b    = bh / NH;
  const int h    = bh % NH;
  const int t    = threadIdx.x;
  const int wave = t >> 5;
  const int lane = t & 31;
  const int lo   = lane & 15;
  const int hi   = lane >> 4;
  const int q0   = blockIdx.y * 128 + wave * 16;

  const f16* Qh = Q + (size_t)bh * SEQ * HD;
  const f16* Kh = K + (size_t)bh * SEQ * HD;
  const f16* Vh = V + (size_t)bh * SEQ * HD;       // transposed [d][tok]
  const float* mrow = mask + (size_t)b * SEQ;
  f16* Pw = &Pt[wave * 16 * KT];

  v16h qa[2];
  {
    int row = q0 + lo;
    int kk  = hi * 16;
    #pragma unroll
    for (int c = 0; c < 2; ++c) {
      const f16* p = Qh + (size_t)row * HD + c * 32 + kk;
      qa[c] = cat8(*(const v8h*)p, *(const v8h*)(p + 8));
    }
  }

  v8f o[4] = {};
  float mrun[8], lrun[8];
  #pragma unroll
  for (int r = 0; r < 8; ++r) { mrun[r] = -1e30f; lrun[r] = 0.0f; }

  if (wave == 0) {
    tdm_load_2d(lds_off_of(&Ks[0][0]), Kh, HD, KT, HD, 1);     // 64 toks x 64 d
    tdm_load_2d(lds_off_of(&Vt[0][0]), Vh, KT, HD, SEQ, 1);    // 64 d rows x 64 toks
    __builtin_amdgcn_s_wait_tensorcnt(0);
  }
  __syncthreads();

  int cur = 0;
  for (int kt = 0; kt < SEQ; kt += KT, cur ^= 1) {
    if (wave == 0 && kt + KT < SEQ) {
      tdm_load_2d(lds_off_of(&Ks[cur ^ 1][0]), &Kh[(size_t)(kt + KT) * HD], HD, KT, HD, 1);
      tdm_load_2d(lds_off_of(&Vt[cur ^ 1][0]), &Vh[kt + KT], KT, HD, SEQ, 1);
    }

    // ---- scores: S[16 x 64] = Q(16x64) @ K^T ----
    v8f s[4] = {};
    #pragma unroll
    for (int c = 0; c < 2; ++c) {
      v16h bf[4];
      #pragma unroll
      for (int j = 0; j < 4; ++j) {
        const f16* kp = &Ks[cur][(j * 16 + lo) * KS + c * 32 + hi * 16];
        bf[j] = cat8(*(const v8h*)kp, *(const v8h*)(kp + 8));
      }
      #pragma unroll
      for (int j = 0; j < 4; ++j) s[j] = wmma16(qa[c], bf[j], s[j]);
    }

    const float sc = 0.125f;                   // 1/sqrt(64)
    float msk[4];
    #pragma unroll
    for (int j = 0; j < 4; ++j)
      msk[j] = (1.0f - mrow[kt + j * 16 + lo]) * -10000.0f;

    float p[4][8], scl[8];
    #pragma unroll
    for (int r = 0; r < 8; ++r) {
      float a0 = s[0][r] * sc + msk[0];
      float a1 = s[1][r] * sc + msk[1];
      float a2 = s[2][r] * sc + msk[2];
      float a3 = s[3][r] * sc + msk[3];
      float rm = fmaxf(fmaxf(a0, a1), fmaxf(a2, a3));
      #pragma unroll
      for (int off = 1; off < 16; off <<= 1) rm = fmaxf(rm, __shfl_xor(rm, off));
      float mnew = fmaxf(mrun[r], rm);
      float e0 = __expf(a0 - mnew);
      float e1 = __expf(a1 - mnew);
      float e2 = __expf(a2 - mnew);
      float e3 = __expf(a3 - mnew);
      float rs = (e0 + e1) + (e2 + e3);
      #pragma unroll
      for (int off = 1; off < 16; off <<= 1) rs += __shfl_xor(rs, off);
      scl[r]  = __expf(mrun[r] - mnew);
      lrun[r] = lrun[r] * scl[r] + rs;
      mrun[r] = mnew;
      p[0][r] = e0; p[1][r] = e1; p[2][r] = e2; p[3][r] = e3;
    }
    #pragma unroll
    for (int t2 = 0; t2 < 4; ++t2)
      #pragma unroll
      for (int r = 0; r < 8; ++r) o[t2][r] *= scl[r];

    // P (C-layout) -> per-wave LDS row-major [16 x 64]
    #pragma unroll
    for (int j = 0; j < 4; ++j)
      #pragma unroll
      for (int r = 0; r < 8; ++r)
        Pw[(r + 8 * hi) * KT + j * 16 + lo] = (f16)p[j][r];

    // ---- O += P @ V : two 32-token halves ----
    #pragma unroll
    for (int half = 0; half < 2; ++half) {
      const f16* pp = &Pw[lo * KT + half * 32 + hi * 16];
      v16h pa = cat8(*(const v8h*)pp, *(const v8h*)(pp + 8));
      v16h vb[4];
      #pragma unroll
      for (int t2 = 0; t2 < 4; ++t2) {
        const f16* vp = &Vt[cur][(t2 * 16 + lo) * KS + half * 32 + hi * 16];
        vb[t2] = cat8(*(const v8h*)vp, *(const v8h*)(vp + 8));
      }
      #pragma unroll
      for (int t2 = 0; t2 < 4; ++t2) o[t2] = wmma16(pa, vb[t2], o[t2]);
    }

    if (wave == 0) __builtin_amdgcn_s_wait_tensorcnt(0);
    __syncthreads();   // next buffers ready; current buffers free
  }

  #pragma unroll
  for (int t2 = 0; t2 < 4; ++t2)
    #pragma unroll
    for (int r = 0; r < 8; ++r) {
      int row = q0 + r + 8 * hi;
      int d   = t2 * 16 + lo;
      float v = o[t2][r] / lrun[r];
      ctx[((size_t)b * SEQ + row) * HID + h * HD + d] = (f16)v;
    }
}

// ---------------------------------------------------------------- LayerNorm
__global__ void ln_kernel(const float* __restrict__ resid, const float* __restrict__ gamma,
                          const float* __restrict__ beta, float* __restrict__ out) {
  __shared__ float r1[256];
  __shared__ float r2[256];
  const int row = blockIdx.x;
  const int t   = threadIdx.x;
  const float* x = resid + (size_t)row * HID;
  float s = 0.0f, ss = 0.0f;
  for (int i = t; i < HID; i += 256) { float v = x[i]; s += v; ss += v * v; }
  r1[t] = s; r2[t] = ss;
  __syncthreads();
  for (int o2 = 128; o2 > 0; o2 >>= 1) {
    if (t < o2) { r1[t] += r1[t + o2]; r2[t] += r2[t + o2]; }
    __syncthreads();
  }
  const float mean = r1[0] * (1.0f / HID);
  const float var  = r2[0] * (1.0f / HID) - mean * mean;
  const float inv  = rsqrtf(var + 1e-5f);
  for (int i = t; i < HID; i += 256)
    out[(size_t)row * HID + i] = (x[i] - mean) * inv * gamma[i] + beta[i];
}

// ---------------------------------------------------------------- launcher
extern "C" void kernel_launch(void* const* d_in, const int* in_sizes, int n_in,
                              void* d_out, int out_size, void* d_ws, size_t ws_size,
                              hipStream_t stream) {
  (void)in_sizes; (void)n_in; (void)out_size; (void)ws_size;

  const float* hidden = (const float*)d_in[0];
  const float* mask   = (const float*)d_in[1];
  const float* Wq = (const float*)d_in[2];
  const float* bq = (const float*)d_in[3];
  const float* Wk = (const float*)d_in[4];
  const float* bk = (const float*)d_in[5];
  const float* Wv = (const float*)d_in[6];
  const float* bv = (const float*)d_in[7];
  const float* Wo = (const float*)d_in[8];
  const float* bo = (const float*)d_in[9];
  const float* gamma = (const float*)d_in[10];
  const float* beta  = (const float*)d_in[11];
  float* out = (float*)d_out;

  char* ws = (char*)d_ws;
  size_t off = 0;
  const size_t WE = (size_t)HID * HID;
  f16* Xh    = (f16*)(ws + off); off += XE * 2;
  f16* Wt3   = (f16*)(ws + off); off += 3 * WE * 2;      // Q|K|V weights, transposed [2304][768]
  f16* Woth  = (f16*)(ws + off); off += WE * 2;          // Wo transposed [768][768]
  f16* QKVh  = (f16*)(ws + off); off += 3 * XE * 2;      // Q,K head-split; V head-split transposed
  f16* Ctxh  = (f16*)(ws + off); off += XE * 2;
  float* resid = (float*)(ws + off); off += XE * 4;

  cvt_f32_to_f16<<<2048, 256, 0, stream>>>(hidden, Xh, (int)XE);

  dim3 tb(32, 8);
  dim3 tg(HID / 32, HID / 32);
  transpose_w<<<tg, tb, 0, stream>>>(Wq, Wt3, 0);
  transpose_w<<<tg, tb, 0, stream>>>(Wk, Wt3, HID);
  transpose_w<<<tg, tb, 0, stream>>>(Wv, Wt3, 2 * HID);
  transpose_w<<<tg, tb, 0, stream>>>(Wo, Woth, 0);

  gemm768<0><<<dim3(MTOT / 128, (3 * HID) / 64), 256, 0, stream>>>(
      Xh, Wt3, bq, bk, bv, nullptr, QKVh, nullptr);

  attn_kernel<<<dim3(BATCH * NH, SEQ / 128), 256, 0, stream>>>(
      QKVh, QKVh + XE, QKVh + 2 * XE, mask, Ctxh);

  gemm768<1><<<dim3(MTOT / 128, HID / 64), 256, 0, stream>>>(
      Ctxh, Woth, bo, nullptr, nullptr, hidden, nullptr, resid);

  ln_kernel<<<MTOT, 256, 0, stream>>>(resid, gamma, beta, out);
}
